// MoeMLPForLoopDebug_21483426414711
// MI455X (gfx1250) — matrix operand order repaired and checked by reading, not compiled, not run on hardware
//
#include <hip/hip_runtime.h>
#include <math.h>

// Problem constants (match reference)
#define N_TOK   8192          // 4 * 2048 tokens
#define D_EMBD  1024
#define D_FF    4096
#define N_EXP   8
#define MT      16            // token tile (one WMMA M-tile)

typedef __attribute__((ext_vector_type(16))) __bf16 v16bf;
typedef __attribute__((ext_vector_type(8)))  float  v8f;

__device__ __forceinline__ unsigned short f2bf(float f) {
    union { float f; unsigned u; } v; v.f = f;
    unsigned r = v.u + 0x7FFFu + ((v.u >> 16) & 1u);   // round-to-nearest-even
    return (unsigned short)(r >> 16);
}

// ---------------------------------------------------------------------------
// Kernel 1: zero output, convert x (f32 -> bf16) into ws, zero expert counters
// ---------------------------------------------------------------------------
__global__ void moe_init_kernel(const float* __restrict__ x,
                                unsigned short* __restrict__ xbf,
                                float* __restrict__ out,
                                int* __restrict__ counts) {
    int tid = blockIdx.x * blockDim.x + threadIdx.x;
    const int total = N_TOK * D_EMBD;
    for (int i = tid; i < total; i += gridDim.x * blockDim.x) {
        xbf[i] = f2bf(x[i]);
        out[i] = 0.0f;
    }
    if (tid < N_EXP) counts[tid] = 0;
}

// ---------------------------------------------------------------------------
// Kernel 2: router. One wave32 per token: logits, softmax, top-2, renorm,
// write dense gates[N][E] and compacted per-expert token lists.
// ---------------------------------------------------------------------------
__global__ void moe_router_kernel(const float* __restrict__ x,
                                  const float* __restrict__ rw,     // [D][E]
                                  float* __restrict__ gates,        // [N][E]
                                  int* __restrict__ counts,         // [E]
                                  int* __restrict__ lists) {        // [E][N]
    const int wave = threadIdx.x >> 5;
    const int lane = threadIdx.x & 31;
    const int t = blockIdx.x * (blockDim.x >> 5) + wave;
    if (t >= N_TOK) return;

    const float* xr = x + (size_t)t * D_EMBD;
    float acc[N_EXP] = {};
    for (int d = lane; d < D_EMBD; d += 32) {
        float xv = xr[d];
        const float4* w4 = (const float4*)(rw + (size_t)d * N_EXP);
        float4 a = w4[0], b = w4[1];
        acc[0] += xv * a.x; acc[1] += xv * a.y; acc[2] += xv * a.z; acc[3] += xv * a.w;
        acc[4] += xv * b.x; acc[5] += xv * b.y; acc[6] += xv * b.z; acc[7] += xv * b.w;
    }
    #pragma unroll
    for (int off = 16; off > 0; off >>= 1) {
        #pragma unroll
        for (int e = 0; e < N_EXP; ++e) acc[e] += __shfl_xor(acc[e], off, 32);
    }
    if (lane == 0) {
        float mx = acc[0];
        #pragma unroll
        for (int e = 1; e < N_EXP; ++e) mx = fmaxf(mx, acc[e]);
        float p[N_EXP];
        #pragma unroll
        for (int e = 0; e < N_EXP; ++e) p[e] = expf(acc[e] - mx);
        int i1 = 0;
        #pragma unroll
        for (int e = 1; e < N_EXP; ++e) if (p[e] > p[i1]) i1 = e;
        int i2 = (i1 == 0) ? 1 : 0;
        #pragma unroll
        for (int e = 0; e < N_EXP; ++e) if (e != i1 && p[e] > p[i2]) i2 = e;
        float g1 = p[i1], g2 = p[i2];
        float inv = 1.0f / (g1 + g2);           // norm_topk_prob (softmax denom cancels)
        g1 *= inv; g2 *= inv;
        #pragma unroll
        for (int e = 0; e < N_EXP; ++e) gates[(size_t)t * N_EXP + e] = 0.0f;
        gates[(size_t)t * N_EXP + i1] = g1;
        gates[(size_t)t * N_EXP + i2] = g2;
        int p1 = atomicAdd(&counts[i1], 1); lists[(size_t)i1 * N_TOK + p1] = t;
        int p2 = atomicAdd(&counts[i2], 1); lists[(size_t)i2 * N_TOK + p2] = t;
    }
}

// ---------------------------------------------------------------------------
// Kernel 2.5: tiled transpose + f32->bf16 convert of per-expert weights.
// src: [E][R][C] f32  ->  dst: [E][C][R] bf16.  Runs once per launch.
// ---------------------------------------------------------------------------
__global__ void moe_transpose_kernel(const float* __restrict__ src,
                                     unsigned short* __restrict__ dst,
                                     int R, int C) {
    __shared__ float tile[32][33];
    const int e  = blockIdx.z;
    const int c0 = blockIdx.x * 32;
    const int r0 = blockIdx.y * 32;
    const int tx = threadIdx.x & 31;
    const int ty = threadIdx.x >> 5;          // 8 rows per pass
    const float* s = src + (size_t)e * R * C;
    unsigned short* d = dst + (size_t)e * C * R;
    #pragma unroll
    for (int i = 0; i < 4; ++i) {
        int r = ty + i * 8;
        tile[r][tx] = s[(size_t)(r0 + r) * C + c0 + tx];      // coalesced read
    }
    __syncthreads();
    #pragma unroll
    for (int i = 0; i < 4; ++i) {
        int r = ty + i * 8;
        d[(size_t)(c0 + r) * R + r0 + tx] = f2bf(tile[tx][r]); // coalesced write
    }
}

// ---------------------------------------------------------------------------
// WMMA fragment loaders from LDS (bf16, wave32 layouts per CDNA5 ISA 7.12.2)
// A 16x32: lane m=lane&15, K runs [koff,koff+8) and [koff+16,koff+24), koff=8*(lane>=16)
// B 32x16: lane n=lane&15, K runs [koff,koff+8) and [koff+16,koff+24), koff=8*(lane>=16)
// ---------------------------------------------------------------------------
__device__ __forceinline__ v16bf fragA(const unsigned short* base, int stride, int k0, int lane) {
    const unsigned short* p = base + (lane & 15) * stride + k0 + ((lane >> 4) << 3);
    union { v16bf v; uint4 q[2]; } u;
    u.q[0] = *(const uint4*)(p);
    u.q[1] = *(const uint4*)(p + 16);
    return u.v;
}
__device__ __forceinline__ v16bf fragB(const unsigned short* base, int stride, int k0, int lane) {
    const unsigned short* p = base + (lane & 15) * stride + k0 + ((lane >> 4) << 3);
    union { v16bf v; uint4 q[2]; } u;
    u.q[0] = *(const uint4*)(p);
    u.q[1] = *(const uint4*)(p + 16);
    return u.v;
}

// ---------------------------------------------------------------------------
// Kernel 3: fused expert MLP. Block = (expert e, 16-token tile from its list).
// Entire gelu(h) tile [16 x 4096] bf16 stays in LDS (CDNA5: 320KB per WGP).
// 8 waves split output columns. PRET=true: weight panels are pre-transposed
// bf16 in ws -> staging is a pure b128 copy. PRET=false: convert f32 in-block.
// ---------------------------------------------------------------------------
template <bool PRET>
__global__ __launch_bounds__(256, 1)
void moe_mlp_kernel(const unsigned short* __restrict__ xbf,
                    const float* __restrict__ wfc,   // [E][D][F] f32
                    const float* __restrict__ bfc,   // [E][F]
                    const float* __restrict__ wproj, // [E][F][D] f32
                    const float* __restrict__ bproj, // [E][D]
                    const unsigned short* __restrict__ w1t, // [E][F][D] bf16 (PRET)
                    const unsigned short* __restrict__ w2t, // [E][D][F] bf16 (PRET)
                    const float* __restrict__ gates, // [N][E]
                    const int* __restrict__ counts,  // [E]
                    const int* __restrict__ lists,   // [E][N]
                    float* __restrict__ out) {       // [N][D]
    // row pads chosen so row stride mod 64 dwords == 4 -> conflict-spread frags
    __shared__ unsigned short xa[MT][D_EMBD + 8];      //  33.0 KB
    __shared__ unsigned short hs[MT][D_FF + 8];        // 131.3 KB
    __shared__ unsigned short wst[8][MT][128 + 8];     //  34.8 KB per-wave W^T panels
    __shared__ int   tok_s[MT];
    __shared__ float gate_s[MT];

    const int e    = blockIdx.y;
    const int cnt  = counts[e];
    const int row0 = blockIdx.x * MT;
    if (row0 >= cnt) return;

    const int tid  = threadIdx.x;
    const int wave = tid >> 5;
    const int lane = tid & 31;

    if (tid < MT) {
        int r  = row0 + tid;
        int tk = lists[(size_t)e * N_TOK + (r < cnt ? r : 0)];
        tok_s[tid]  = tk;
        gate_s[tid] = (r < cnt) ? gates[(size_t)tk * N_EXP + e] : 0.0f;  // ghost rows: gate 0
    }
    __syncthreads();

    // cooperative load of 16 gathered token rows into xa (bf16, 128B chunks)
    {
        int m = tid >> 4, c = tid & 15;
        const uint4* src = (const uint4*)(xbf + (size_t)tok_s[m] * D_EMBD + c * 64);
        uint4* dst = (uint4*)(&xa[m][c * 64]);
        #pragma unroll
        for (int j = 0; j < 8; ++j) dst[j] = src[j];
    }
    __syncthreads();

    unsigned short (*wstw)[136] = wst[wave];
    const int srow = lane >> 1;             // staging row (PRET copy path)
    const int shlf = (lane & 1) * 64;       // staging half-row offset (elems)

    // ---------------- Phase A: h = gelu(x @ W1 + b1), wave owns 512 cols ----
    const float* wfce = wfc + (size_t)e * D_EMBD * D_FF;
    const float* bfce = bfc + (size_t)e * D_FF;
    const unsigned short* w1te = w1t + (size_t)e * D_FF * D_EMBD;
    for (int t = 0; t < 32; ++t) {
        const int n0 = wave * 512 + t * 16;
        v8f acc = {};
        for (int kc = 0; kc < 8; ++kc) {
            const int k0 = kc * 128;
            if (PRET) {
                // pure bf16 copy: W1^T[n0+srow][k0+shlf .. +32]  (64B per lane)
                const unsigned short* sp = w1te + (size_t)(n0 + srow) * D_EMBD + k0 + shlf;
                if (kc < 7) __builtin_prefetch(sp + 128, 0, 3);
                const uint4* s4 = (const uint4*)sp;
                uint4* d4 = (uint4*)(&wstw[srow][shlf]);
                #pragma unroll
                for (int j = 0; j < 4; ++j) d4[j] = s4[j];
            } else {
                // stage W1[k0:k0+128, n0:n0+16] transposed+converted
                #pragma unroll
                for (int rep = 0; rep < 4; ++rep) {
                    const int kl = rep * 32 + lane;
                    const float4* w4 = (const float4*)(wfce + (size_t)(k0 + kl) * D_FF + n0);
                    if (kc < 7 && rep == 0)
                        __builtin_prefetch(wfce + (size_t)(k0 + 128 + kl) * D_FF + n0, 0, 3);
                    #pragma unroll
                    for (int j4 = 0; j4 < 4; ++j4) {
                        float4 v = w4[j4];
                        wstw[j4 * 4 + 0][kl] = f2bf(v.x);
                        wstw[j4 * 4 + 1][kl] = f2bf(v.y);
                        wstw[j4 * 4 + 2][kl] = f2bf(v.z);
                        wstw[j4 * 4 + 3][kl] = f2bf(v.w);
                    }
                }
            }
            #pragma unroll
            for (int kk = 0; kk < 4; ++kk) {
                v16bf a = fragA(&xa[0][0], D_EMBD + 8, k0 + kk * 32, lane);
                v16bf b = fragB(&wstw[0][0], 136, kk * 32, lane);
                acc = __builtin_amdgcn_wmma_f32_16x16x32_bf16(
                          false, a, false, b, (short)0, acc, false, false);
            }
        }
        const int n  = lane & 15;
        const int mb = (lane >> 4) << 3;
        const float bias = bfce[n0 + n];
        #pragma unroll
        for (int r = 0; r < 8; ++r) {
            float v = acc[r] + bias;
            v = 0.5f * v * (1.0f + erff(v * 0.70710678118f));   // exact gelu
            hs[mb + r][n0 + n] = f2bf(v);
        }
    }
    __syncthreads();

    // ---------------- Phase B: y = h @ W2 + b2; out += gate * y ------------
    const float* wpre = wproj + (size_t)e * D_FF * D_EMBD;
    const float* bpre = bproj + (size_t)e * D_EMBD;
    const unsigned short* w2te = w2t + (size_t)e * D_EMBD * D_FF;
    for (int t = 0; t < 8; ++t) {
        const int n0 = wave * 128 + t * 16;
        v8f acc = {};
        for (int kc = 0; kc < 32; ++kc) {
            const int k0 = kc * 128;
            if (PRET) {
                const unsigned short* sp = w2te + (size_t)(n0 + srow) * D_FF + k0 + shlf;
                if (kc < 31) __builtin_prefetch(sp + 128, 0, 3);
                const uint4* s4 = (const uint4*)sp;
                uint4* d4 = (uint4*)(&wstw[srow][shlf]);
                #pragma unroll
                for (int j = 0; j < 4; ++j) d4[j] = s4[j];
            } else {
                #pragma unroll
                for (int rep = 0; rep < 4; ++rep) {
                    const int kl = rep * 32 + lane;
                    const float4* w4 = (const float4*)(wpre + (size_t)(k0 + kl) * D_EMBD + n0);
                    if (kc < 31 && rep == 0)
                        __builtin_prefetch(wpre + (size_t)(k0 + 128 + kl) * D_EMBD + n0, 0, 3);
                    #pragma unroll
                    for (int j4 = 0; j4 < 4; ++j4) {
                        float4 v = w4[j4];
                        wstw[j4 * 4 + 0][kl] = f2bf(v.x);
                        wstw[j4 * 4 + 1][kl] = f2bf(v.y);
                        wstw[j4 * 4 + 2][kl] = f2bf(v.z);
                        wstw[j4 * 4 + 3][kl] = f2bf(v.w);
                    }
                }
            }
            #pragma unroll
            for (int kk = 0; kk < 4; ++kk) {
                v16bf a = fragA(&hs[0][0], D_FF + 8, k0 + kk * 32, lane);
                v16bf b = fragB(&wstw[0][0], 136, kk * 32, lane);
                acc = __builtin_amdgcn_wmma_f32_16x16x32_bf16(
                          false, a, false, b, (short)0, acc, false, false);
            }
        }
        const int n  = lane & 15;
        const int mb = (lane >> 4) << 3;
        const float bias = bpre[n0 + n];
        #pragma unroll
        for (int r = 0; r < 8; ++r) {
            const int m = mb + r;
            float v = (acc[r] + bias) * gate_s[m];
            atomicAdd(&out[(size_t)tok_s[m] * D_EMBD + n0 + n], v);
        }
    }
}

// ---------------------------------------------------------------------------
extern "C" void kernel_launch(void* const* d_in, const int* in_sizes, int n_in,
                              void* d_out, int out_size, void* d_ws, size_t ws_size,
                              hipStream_t stream) {
    (void)in_sizes; (void)n_in; (void)out_size;
    const float* x     = (const float*)d_in[0];
    const float* rw    = (const float*)d_in[1];
    const float* wfc   = (const float*)d_in[2];
    const float* bfc   = (const float*)d_in[3];
    const float* wproj = (const float*)d_in[4];
    const float* bproj = (const float*)d_in[5];
    float* out = (float*)d_out;

    // workspace layout
    const size_t SZ_XBF   = (size_t)N_TOK * D_EMBD * 2;        // 16 MB
    const size_t SZ_GATES = (size_t)N_TOK * N_EXP * 4;         // 256 KB
    const size_t SZ_CNT   = 256;
    const size_t SZ_LISTS = (size_t)N_EXP * N_TOK * 4;         // 256 KB
    const size_t SZ_WT    = (size_t)N_EXP * D_EMBD * D_FF * 2; // 64 MB each

    char* ws = (char*)d_ws;
    unsigned short* xbf = (unsigned short*)ws;   ws += SZ_XBF;
    float* gates        = (float*)ws;            ws += SZ_GATES;
    int*   counts       = (int*)ws;              ws += SZ_CNT;
    int*   lists        = (int*)ws;              ws += SZ_LISTS;
    unsigned short* w1t = (unsigned short*)ws;   ws += SZ_WT;
    unsigned short* w2t = (unsigned short*)ws;

    const size_t need = SZ_XBF + SZ_GATES + SZ_CNT + SZ_LISTS + 2 * SZ_WT; // ~152 MB
    const bool pret = (ws_size >= need);   // constant per process -> deterministic

    moe_init_kernel<<<4096, 256, 0, stream>>>(x, xbf, out, counts);
    moe_router_kernel<<<N_TOK / 8, 256, 0, stream>>>(x, rw, gates, counts, lists);

    if (pret) {
        // W1: [E][D][F] -> [E][F][D] bf16 ; W2: [E][F][D] -> [E][D][F] bf16
        moe_transpose_kernel<<<dim3(D_FF / 32, D_EMBD / 32, N_EXP), 256, 0, stream>>>(
            wfc, w1t, D_EMBD, D_FF);
        moe_transpose_kernel<<<dim3(D_EMBD / 32, D_FF / 32, N_EXP), 256, 0, stream>>>(
            wproj, w2t, D_FF, D_EMBD);
        moe_mlp_kernel<true><<<dim3(N_TOK / MT, N_EXP), 256, 0, stream>>>(
            xbf, wfc, bfc, wproj, bproj, w1t, w2t, gates, counts, lists, out);
    } else {
        moe_mlp_kernel<false><<<dim3(N_TOK / MT, N_EXP), 256, 0, stream>>>(
            xbf, wfc, bfc, wproj, bproj, w1t, w2t, gates, counts, lists, out);
    }
}
